// OIMLossSafeNew_9105330668001
// MI455X (gfx1250) — matrix operand order repaired and checked by reading, not compiled, not run on hardware
//
#include <hip/hip_runtime.h>
#include <hip/hip_bf16.h>
#include <math.h>

// ---------------------------------------------------------------------------
// OIM loss forward + LUT/CQ update for MI455X (gfx1250, wave32, WMMA bf16)
// ---------------------------------------------------------------------------

typedef __attribute__((ext_vector_type(16))) __bf16 v16bf;
typedef __attribute__((ext_vector_type(8)))  float  v8f;
typedef __attribute__((ext_vector_type(4)))  float  f32x4;

#define F_DIM   256
#define B_ROWS  512
#define P_LUT   50000
#define Q_CQ    5000
#define N_TOT   (P_LUT + Q_CQ)
#define EPSN    1e-12f
#define SCALARF 30.0f

#define BLK_N   128
#define BLK_M   64
#define KSTEP   32
#define NKSTEP  (F_DIM / KSTEP)                 // 8
#define NBLK_N  ((N_TOT + BLK_N - 1) / BLK_N)   // 430
#define NCOLS2  (NBLK_N * 2)                    // 860 wave-column partials
#define NEGBIG  -3.0e38f
#define LDS_RS  40                              // LDS row stride (bf16)

// ---------------------------------------------------------------------------
__global__ void oim_init_kernel(float* cssl, float* cssu) {
    int f = threadIdx.x;
    if (f < F_DIM) { cssl[f] = 0.0f; cssu[f] = 0.0f; }
}

// Fused: column sum-of-squares (per-feature norm of W.T rows) + copy src rows
// into the output buffer (new_lut / new_cq initial value).
__global__ void oim_colpass_kernel(const float* __restrict__ src, int rows,
                                   float* __restrict__ colss,
                                   float* __restrict__ dst) {
    int f  = threadIdx.x;           // 256 threads = one feature each (coalesced)
    int r0 = blockIdx.x * 64;
    float acc = 0.0f;
#pragma unroll 4
    for (int r = 0; r < 64; ++r) {
        int row = r0 + r;
        if (row < rows) {
            float v = src[row * F_DIM + f];
            acc += v * v;
            dst[row * F_DIM + f] = v;
        }
    }
    atomicAdd(&colss[f], acc);
}

// Per-row all-zero flag (thread per row, vectorized b128 streaming).
__global__ void oim_badrow_kernel(const float* __restrict__ src, int rows,
                                  int* __restrict__ bad) {
    int r = blockIdx.x * blockDim.x + threadIdx.x;
    if (r >= rows) return;
    const f32x4* p = (const f32x4*)(src + (size_t)r * F_DIM);
    int nz = 0;
#pragma unroll 8
    for (int i = 0; i < F_DIM / 4; ++i) {
        f32x4 v = p[i];
        nz |= (v.x != 0.0f) | (v.y != 0.0f) | (v.z != 0.0f) | (v.w != 0.0f);
    }
    bad[r] = nz ? 0 : 1;
}

__global__ void oim_scales_kernel(const float* cssl, const float* cssu,
                                  float* sl, float* su) {
    int f = threadIdx.x;
    if (f < F_DIM) {
        sl[f] = 1.0f / fmaxf(sqrtf(cssl[f]), EPSN);
        su[f] = 1.0f / fmaxf(sqrtf(cssu[f]), EPSN);
    }
}

__device__ __forceinline__ float block_reduce_sum_256(float v, float* sm) {
    int t = threadIdx.x;
    sm[t] = v;
    __syncthreads();
#pragma unroll
    for (int s = 128; s > 0; s >>= 1) {
        if (t < s) sm[t] += sm[t + s];
        __syncthreads();
    }
    float r = sm[0];
    __syncthreads();
    return r;
}

// Normalize each input row and fold the per-feature 1/colnorm scales into two
// bf16 A matrices: A_l (for lut columns) and A_u (for cq columns).
__global__ void oim_anorm_kernel(const float* __restrict__ inputs,
                                 const float* __restrict__ s_l,
                                 const float* __restrict__ s_u,
                                 __bf16* __restrict__ Al,
                                 __bf16* __restrict__ Au) {
    __shared__ float sm[256];
    int m = blockIdx.x, f = threadIdx.x;
    float v  = inputs[m * F_DIM + f];
    float ss = block_reduce_sum_256(v * v, sm);
    float a  = v / fmaxf(sqrtf(ss), EPSN);
    Al[m * F_DIM + f] = (__bf16)(a * s_l[f]);
    Au[m * F_DIM + f] = (__bf16)(a * s_u[f]);
}

// ---------------------------------------------------------------------------
// GEMM main loop: LDS double-buffered B tiles + bf16 WMMA. DUAL=true only for
// the rare block that straddles the lut/cq boundary (needs per-tile A select).
// ---------------------------------------------------------------------------
__device__ __forceinline__ void oim_stage_store(__bf16* base, int idx,
                                                f32x4 v0, f32x4 v1, float mk) {
    v0 = v0 * mk;
    v1 = v1 * mk;
    union { __bf16 h[8]; f32x4 q; } o;
    o.h[0] = (__bf16)v0.x; o.h[1] = (__bf16)v0.y;
    o.h[2] = (__bf16)v0.z; o.h[3] = (__bf16)v0.w;
    o.h[4] = (__bf16)v1.x; o.h[5] = (__bf16)v1.y;
    o.h[6] = (__bf16)v1.z; o.h[7] = (__bf16)v1.w;
    *(f32x4*)(base + idx) = o.q;               // ds_store_b128
}

template <bool DUAL>
__device__ __forceinline__ void oim_mainloop(
    const float* const rowBase[2], const float rowMask[2], const int ldsIdx[2],
    __bf16* BsFlat, const __bf16* A0, const __bf16* A1,
    size_t aOff, int koff, int fragRow0, const bool useL[4], v8f acc[4]) {

    // ---- prologue: stage tile k=0 into buffer 0 (b128 LDS stores) ----
#pragma unroll
    for (int i = 0; i < 2; ++i) {
        f32x4 v0 = *(const f32x4*)(rowBase[i]);
        f32x4 v1 = *(const f32x4*)(rowBase[i] + 4);
        oim_stage_store(BsFlat, ldsIdx[i], v0, v1, rowMask[i]);
    }
    __syncthreads();

#pragma unroll
    for (int ks = 0; ks < NKSTEP; ++ks) {
        const int k0  = ks * KSTEP;
        const int buf = ks & 1;

        // issue next tile's global loads early (double-buffer overlap)
        f32x4 nv0[2], nv1[2];
        if (ks + 1 < NKSTEP) {
#pragma unroll
            for (int i = 0; i < 2; ++i) {
                nv0[i] = *(const f32x4*)(rowBase[i] + (k0 + KSTEP));
                nv1[i] = *(const f32x4*)(rowBase[i] + (k0 + KSTEP) + 4);
                if (ks + 2 < NKSTEP)
                    __builtin_prefetch(rowBase[i] + (k0 + 2 * KSTEP), 0, 0);
            }
        }

        // A fragment(s)
        union { v16bf v; f32x4 f[2]; } uaL, uaU;
        uaL.f[0] = *(const f32x4*)(A0 + aOff + k0);
        uaL.f[1] = *(const f32x4*)(A0 + aOff + k0 + 16);
        if (DUAL) {
            uaU.f[0] = *(const f32x4*)(A1 + aOff + k0);
            uaU.f[1] = *(const f32x4*)(A1 + aOff + k0 + 16);
        }

        // B fragments (32x16): lane n = tile+ln, k = 8*lhalf + h (h<8), 16 + 8*lhalf + (h-8) (h>=8)
#pragma unroll
        for (int t = 0; t < 4; ++t) {
            const __bf16* Bp = BsFlat + buf * (BLK_N * LDS_RS)
                             + (fragRow0 + t * 16) * LDS_RS + koff;
            union { v16bf v; f32x4 f[2]; } ub;
            ub.f[0] = *(const f32x4*)(Bp);
            ub.f[1] = *(const f32x4*)(Bp + 16);
            v16bf af = (DUAL && !useL[t]) ? uaU.v : uaL.v;
            acc[t] = __builtin_amdgcn_wmma_f32_16x16x32_bf16(
                false, af, false, ub.v, (short)0, acc[t], false, false);
        }

        // store next tile into the other buffer
        if (ks + 1 < NKSTEP) {
            __bf16* base = BsFlat + (buf ^ 1) * (BLK_N * LDS_RS);
#pragma unroll
            for (int i = 0; i < 2; ++i)
                oim_stage_store(base, ldsIdx[i], nv0[i], nv1[i], rowMask[i]);
        }
        __syncthreads();
    }
}

// Main GEMM: C[512, 55000] = A_norm @ [lut|cq]^T, fused online-softmax.
// Block = 256 threads = 8 waves arranged 4(M) x 2(N); block tile 64M x 128N.
__global__ void __launch_bounds__(256)
oim_gemm_kernel(const float* __restrict__ lut, const float* __restrict__ cq,
                const __bf16* __restrict__ Al, const __bf16* __restrict__ Au,
                const int* __restrict__ bad_l, const int* __restrict__ bad_u,
                const int* __restrict__ label,
                float* __restrict__ pmax, float* __restrict__ psum,
                float* __restrict__ lblLogit) {
    __shared__ __bf16 Bs[2][BLK_N][LDS_RS];  // 32 k + 8 pad (80B rows)

    const int tid  = threadIdx.x;
    const int lane = tid & 31;
    const int wave = tid >> 5;
    const int wm   = wave >> 1;        // 0..3 (M)
    const int wn   = wave & 1;         // 0..1 (N)
    const int m0    = blockIdx.y * BLK_M + wm * 16;
    const int nblk0 = blockIdx.x * BLK_N;
    const int ln    = lane & 15;
    const int lhalf = lane >> 4;

    // ---- k-invariant staging descriptors: 512 groups of 8 floats ----
    const float* rowBase[2];
    float rowMask[2];
    int   ldsIdx[2];
#pragma unroll
    for (int i = 0; i < 2; ++i) {
        int slot = tid + i * 256;          // 0..511: nl = slot/4, f8 = slot%4
        int nl = slot >> 2, f8 = slot & 3;
        int n  = nblk0 + nl;
        int nc = (n < N_TOT) ? n : (N_TOT - 1);
        rowBase[i] = (nc < P_LUT) ? (lut + (size_t)nc * F_DIM + f8 * 8)
                                  : (cq + (size_t)(nc - P_LUT) * F_DIM + f8 * 8);
        rowMask[i] = (n < N_TOT) ? 1.0f : 0.0f;
        ldsIdx[i]  = nl * LDS_RS + f8 * 8;
    }
    __bf16* BsFlat = &Bs[0][0][0];

    // A fragment addressing: lane m = m0+ln; halves h0..7 -> k = 8*lhalf + h,
    // h8..15 -> +16 (per documented 16-bit A 16x32 layout).
    const size_t aOff = (size_t)(m0 + ln) * F_DIM + (lhalf << 3);
    const int    koff = lhalf << 3;
    const int fragRow0 = wn * 64 + ln;

    v8f acc[4];
#pragma unroll
    for (int t = 0; t < 4; ++t) acc[t] = (v8f)0.0f;

    // Wave-uniform A-side selection per 16-col tile (50000 % 16 == 0).
    bool useL[4];
#pragma unroll
    for (int t = 0; t < 4; ++t)
        useL[t] = (nblk0 + wn * 64 + t * 16) < P_LUT;

    // Block-uniform fast paths: only blockIdx.x == P_LUT/BLK_N mixes sides.
    if (nblk0 + BLK_N <= P_LUT) {
        oim_mainloop<false>(rowBase, rowMask, ldsIdx, BsFlat, Al, Al,
                            aOff, koff, fragRow0, useL, acc);
    } else if (nblk0 >= P_LUT) {
        oim_mainloop<false>(rowBase, rowMask, ldsIdx, BsFlat, Au, Au,
                            aOff, koff, fragRow0, useL, acc);
    } else {
        oim_mainloop<true>(rowBase, rowMask, ldsIdx, BsFlat, Al, Au,
                           aOff, koff, fragRow0, useL, acc);
    }

    // ---- epilogue: fixups, x30, per-row online-softmax partial over 64 cols
    for (int r = 0; r < 8; ++r) {
        int m   = m0 + r + lhalf * 8;          // C layout: vgpr r, lane half
        int y   = label[m];
        int lcl = min(max(y, 0), N_TOT - 1);
        float proj[4];
#pragma unroll
        for (int t = 0; t < 4; ++t) {
            int n = nblk0 + wn * 64 + t * 16 + ln;
            float val = NEGBIG;
            if (n < N_TOT) {
                float v  = acc[t][r];
                int bad  = (n < P_LUT) ? bad_l[n] : bad_u[n - P_LUT];
                float o  = bad ? -1.0f : v;
                if (n < P_LUT && y < P_LUT && y == n && bad) o = 1.0f;
                o *= SCALARF;
                if (n == lcl) lblLogit[m] = o;
                val = o;
            }
            proj[t] = val;
        }
        float mx = fmaxf(fmaxf(proj[0], proj[1]), fmaxf(proj[2], proj[3]));
#pragma unroll
        for (int o = 1; o < 16; o <<= 1) mx = fmaxf(mx, __shfl_xor(mx, o, 32));
        float s = 0.0f;
#pragma unroll
        for (int t = 0; t < 4; ++t)
            s += (proj[t] <= -1e37f) ? 0.0f : __expf(proj[t] - mx);
#pragma unroll
        for (int o = 1; o < 16; o <<= 1) s += __shfl_xor(s, o, 32);
        if (ln == 0) {
            int j2 = blockIdx.x * 2 + wn;
            pmax[(size_t)j2 * B_ROWS + m] = mx;
            psum[(size_t)j2 * B_ROWS + m] = s;
        }
    }
}

// ---------------------------------------------------------------------------
__global__ void oim_finalize_kernel(const float* __restrict__ pmax,
                                    const float* __restrict__ psum,
                                    const float* __restrict__ lbl,
                                    const int* __restrict__ label,
                                    float* __restrict__ out) {
    __shared__ float sm[B_ROWS];
    int m = threadIdx.x;               // 512 threads, one row each
    float M = NEGBIG;
    for (int j = 0; j < NCOLS2; ++j)
        M = fmaxf(M, pmax[(size_t)j * B_ROWS + m]);
    float S = 0.0f;
    for (int j = 0; j < NCOLS2; ++j) {
        float pm = pmax[(size_t)j * B_ROWS + m];
        float ps = psum[(size_t)j * B_ROWS + m];
        S += (pm <= -1e37f) ? 0.0f : ps * __expf(pm - M);
    }
    float per = 0.0f;
    if (label[m] != P_LUT) per = -(lbl[m] - M - __logf(S));
    sm[m] = per;
    __syncthreads();
#pragma unroll
    for (int s = 256; s > 0; s >>= 1) {
        if (m < s) sm[m] += sm[m + s];
        __syncthreads();
    }
    if (m == 0) out[0] = sm[0] / (float)B_ROWS;
}

// Sequential EMA / circular-queue scatter (honors scan semantics; one block).
__global__ void oim_ema_kernel(const float* __restrict__ inputs,
                               const int* __restrict__ label,
                               float* __restrict__ newLut,
                               float* __restrict__ newCq) {
    __shared__ float sm[256];
    int f = threadIdx.x;
    int hdr = 0;
    for (int i = 0; i < B_ROWS; ++i) {
        int y = label[i];
        float x = inputs[i * F_DIM + f];
        if (y < P_LUT) {
            int yc = min(max(y, 0), P_LUT - 1);
            float nx2 = block_reduce_sum_256(x * x, sm);
            float xn  = x / fmaxf(sqrtf(nx2), EPSN);
            float row = newLut[(size_t)yc * F_DIM + f];
            float rr  = 0.5f * row + 0.5f * xn;
            float nr2 = block_reduce_sum_256(rr * rr, sm);
            newLut[(size_t)yc * F_DIM + f] = rr / fmaxf(sqrtf(nr2), EPSN);
            __syncthreads();
        } else {
            newCq[(size_t)(hdr % Q_CQ) * F_DIM + f] = x;
            ++hdr;
        }
    }
}

// ---------------------------------------------------------------------------
extern "C" void kernel_launch(void* const* d_in, const int* in_sizes, int n_in,
                              void* d_out, int out_size, void* d_ws, size_t ws_size,
                              hipStream_t stream) {
    const float* inputs = (const float*)d_in[0];
    const int*   label  = (const int*)d_in[1];
    const float* lut    = (const float*)d_in[2];
    const float* cq     = (const float*)d_in[3];

    float* out    = (float*)d_out;
    float* loss   = out;
    float* newLut = out + 1;
    float* newCq  = out + 1 + (size_t)P_LUT * F_DIM;

    char* ws = (char*)d_ws;
    size_t off = 0;
    auto carve = [&](size_t bytes) {
        void* p = ws + off;
        off = (off + bytes + 255) & ~(size_t)255;
        return p;
    };
    __bf16* Al      = (__bf16*)carve((size_t)B_ROWS * F_DIM * sizeof(__bf16));
    __bf16* Au      = (__bf16*)carve((size_t)B_ROWS * F_DIM * sizeof(__bf16));
    float*  sl      = (float*) carve(F_DIM * sizeof(float));
    float*  su      = (float*) carve(F_DIM * sizeof(float));
    float*  cssl    = (float*) carve(F_DIM * sizeof(float));
    float*  cssu    = (float*) carve(F_DIM * sizeof(float));
    int*    badl    = (int*)   carve((size_t)P_LUT * sizeof(int));
    int*    badu    = (int*)   carve((size_t)Q_CQ * sizeof(int));
    float*  pmax    = (float*) carve((size_t)NCOLS2 * B_ROWS * sizeof(float));
    float*  psum    = (float*) carve((size_t)NCOLS2 * B_ROWS * sizeof(float));
    float*  lblLog  = (float*) carve((size_t)B_ROWS * sizeof(float));

    oim_init_kernel<<<1, 256, 0, stream>>>(cssl, cssu);
    oim_colpass_kernel<<<(P_LUT + 63) / 64, 256, 0, stream>>>(lut, P_LUT, cssl, newLut);
    oim_colpass_kernel<<<(Q_CQ + 63) / 64, 256, 0, stream>>>(cq, Q_CQ, cssu, newCq);
    oim_badrow_kernel<<<(P_LUT + 255) / 256, 256, 0, stream>>>(lut, P_LUT, badl);
    oim_badrow_kernel<<<(Q_CQ + 255) / 256, 256, 0, stream>>>(cq, Q_CQ, badu);
    oim_scales_kernel<<<1, 256, 0, stream>>>(cssl, cssu, sl, su);
    oim_anorm_kernel<<<B_ROWS, 256, 0, stream>>>(inputs, sl, su, Al, Au);
    oim_gemm_kernel<<<dim3(NBLK_N, B_ROWS / BLK_M), 256, 0, stream>>>(
        lut, cq, Al, Au, badl, badu, label, pmax, psum, lblLog);
    oim_finalize_kernel<<<1, B_ROWS, 0, stream>>>(pmax, psum, lblLog, label, loss);
    oim_ema_kernel<<<1, 256, 0, stream>>>(inputs, label, newLut, newCq);
}
